// GriffinResidualBlock_27401891349110
// MI455X (gfx1250) — matrix operand order repaired and checked by reading, not compiled, not run on hardware
//
#include <hip/hip_runtime.h>

// ---------------------------------------------------------------------------
// Griffin residual block for gfx1250 (MI455X).
// Matrix path: bf16 WMMA (v_wmma_f32_16x16x32_bf16), fp32 accumulate.
// Weight tiles fetched by the Tensor Data Mover (tensor_load_to_lds, TENSORcnt).
// ---------------------------------------------------------------------------

#define NTOK   16384      // B*S
#define DMODEL 512
#define DMID   2048       // D*MULT
#define SEQ    2048
#define NBATCH 8
#define KCONV  9

typedef __attribute__((ext_vector_type(16))) __bf16        bf16x16;
typedef __attribute__((ext_vector_type(8)))  float         f32x8;
typedef __attribute__((ext_vector_type(4)))  unsigned int  v4u;
typedef __attribute__((ext_vector_type(8)))  int           v8i;
typedef __attribute__((ext_vector_type(4)))  int           v4i;

union FragAB { uint4 u[2]; bf16x16 v; };

static __device__ __forceinline__ unsigned short f2bf(float f) {
  union { float f; unsigned u; } v; v.f = f;
  unsigned r = v.u + 0x7FFFu + ((v.u >> 16) & 1u);   // round-to-nearest-even
  return (unsigned short)(r >> 16);
}
static __device__ __forceinline__ float gelu_exact(float x) {
  return 0.5f * x * (1.0f + erff(x * 0.70710678118654752f));
}

// epilogue modes
#define MODE_F32        0   // outF = acc + bias
#define MODE_ACCUM      1   // outF += acc
#define MODE_ACCUM_LAST 2   // outF += acc; outB = bf16(outF)
#define MODE_BF16       3   // outB = bf16(acc + bias)
#define MODE_BOTH       4   // outF and outB = acc + bias
#define MODE_SIGMOID    5   // outF = sigmoid(acc + bias)
#define MODE_GELUMUL    6   // t = acc+bias; outB = bf16(gelu(t)*t)
#define MODE_RES        7   // outF = acc + bias + res

// ---------------------------------------------------------------------------
// Weight convert + transpose: dst[n*Kd + k] = bf16(src[off + n*sN + k*sK])
// ---------------------------------------------------------------------------
__global__ void k_cvt_t(const float* __restrict__ src, unsigned short* __restrict__ dst,
                        int N, int Kd, long long sN, long long sK, long long off) {
  long long total = (long long)N * Kd;
  for (long long i = (long long)blockIdx.x * blockDim.x + threadIdx.x; i < total;
       i += (long long)gridDim.x * blockDim.x) {
    long long n = i / Kd, k = i - n * Kd;
    dst[i] = f2bf(src[off + n * sN + k * sK]);
  }
}

// ---------------------------------------------------------------------------
// RMS-ish norm: out = bf16( x / (||x||_2 + eps) * sqrt(D) * g ), per token row
// one block (256 threads) per token, D = 512
// ---------------------------------------------------------------------------
__global__ __launch_bounds__(256) void k_rms(const float* __restrict__ x,
                                             const float* __restrict__ g,
                                             unsigned short* __restrict__ out) {
  __shared__ float red[256];
  const int t = blockIdx.x, tid = threadIdx.x;
  const float* row = x + (long long)t * DMODEL;
  float a = row[tid], b = row[tid + 256];
  red[tid] = a * a + b * b;
  __syncthreads();
  for (int s = 128; s > 0; s >>= 1) {
    if (tid < s) red[tid] += red[tid + s];
    __syncthreads();
  }
  float norm = sqrtf(red[0]);
  float sc = 22.62741699796952f / (norm + 1e-6f);   // sqrt(512)/(||x||+eps)
  unsigned short* orow = out + (long long)t * DMODEL;
  orow[tid]       = f2bf(a * sc * g[tid]);
  orow[tid + 256] = f2bf(b * sc * g[tid + 256]);
}

// ---------------------------------------------------------------------------
// Tiled bf16 WMMA GEMM: C[M,N] = A[M,K] * BT[N,K]^T  (+ epilogue)
// block = 256 threads = 8 waves; tile 128x128, K-step 32.
// Wave (wm = w&3, wn = w>>2) owns a 32(M) x 64(N) subtile = 2x4 WMMA tiles.
// B tile (weights) is DMA'd by the Tensor Data Mover into LDS (wave 0 issues,
// TENSORcnt-tracked); A tile is staged manually because the conv taps need
// mid-tensor per-sequence zero padding.
// ---------------------------------------------------------------------------
__global__ __launch_bounds__(256, 2) void k_gemm(
    const unsigned short* __restrict__ A, const unsigned short* __restrict__ BT,
    const float* __restrict__ bias, float* __restrict__ outF,
    unsigned short* __restrict__ outB, const float* __restrict__ res,
    int M, int N, int Kd, int mode, int shift, int applyShift) {
  __shared__ uint4 lA[128 * 4];   // 128 rows x 32 bf16 (4 x uint4 per row)
  __shared__ uint4 lB[128 * 4];

  const int tid = threadIdx.x;
  const int l   = tid & 31;
  const int w   = tid >> 5;
  const int wm  = w & 3;            // M group: wm*32
  const int wn  = w >> 2;           // N group: wn*64
  const int lm  = l & 15;
  const int lh  = l >> 4;
  const long long mBase = (long long)blockIdx.y * 128;
  const long long nBase = (long long)blockIdx.x * 128;

  // LDS byte address of lB (generic shared pointer: low 32 bits = LDS offset)
  const unsigned ldsB = (unsigned)(unsigned long long)(const void*)&lB[0];

  f32x8 acc[2][4];
#pragma unroll
  for (int mt = 0; mt < 2; ++mt)
#pragma unroll
    for (int nt = 0; nt < 4; ++nt)
#pragma unroll
      for (int v = 0; v < 8; ++v) acc[mt][nt][v] = 0.0f;

  for (int k0 = 0; k0 < Kd; k0 += 32) {
    // ---- B tile: async DMA via Tensor Data Mover (wave 0 issues)
    if (w == 0) {
      const unsigned long long ga =
          (unsigned long long)(const void*)(BT + nBase * (long long)Kd + k0);
      v4u g0; v8i g1;
      v4i g2 = {0, 0, 0, 0}, g3 = {0, 0, 0, 0};
      v8i g4 = {0, 0, 0, 0, 0, 0, 0, 0};
      // group0: count=1 | lds_addr | global_addr[56:0] | type=2
      g0[0] = 1u;
      g0[1] = ldsB;
      g0[2] = (unsigned)(ga & 0xFFFFFFFFull);
      g0[3] = (unsigned)((ga >> 32) & 0x1FFFFFFull) | (2u << 30);
      // group1: data_size=2B; tensor 2D [dim0=Kd, dim1=128 rows];
      //         tile [32 x 128]; dim0 stride = Kd elements
      g1[0] = (1 << 16);                               // data_size = 2 bytes
      g1[1] = (int)((Kd & 0xFFFF) << 16);              // tensor_dim0 lo16
      g1[2] = (int)(((unsigned)Kd >> 16) | (128u << 16)); // dim0 hi16 | dim1 lo16
      g1[3] = (int)(32u << 16);                        // dim1 hi16=0 | tile_dim0=32
      g1[4] = 128;                                     // tile_dim1 = 128
      g1[5] = Kd;                                      // tensor_dim0_stride lo32
      g1[6] = 0;
      g1[7] = 0;
      __builtin_amdgcn_tensor_load_to_lds(g0, g1, g2, g3, g4, 0);
    }

    // ---- A tile (manual: optional conv shift + per-sequence zero padding)
#pragma unroll
    for (int it = 0; it < 2; ++it) {
      const int c = tid + it * 256;          // 512 chunks of 16B per tile
      const int row = c >> 2, kc = c & 3;
      long long t = mBase + row;
      long long sr = t;
      bool valid = true;
      if (applyShift) {
        int s = (int)(t & (SEQ - 1));
        int s2 = s + shift;
        valid = (s2 >= 0) && (s2 < SEQ);
        sr = t + shift;
      }
      uint4 va = make_uint4(0u, 0u, 0u, 0u);
      if (valid) {
        const uint4* pa = (const uint4*)(A + sr * (long long)Kd);
        va = pa[(k0 >> 3) + kc];
        if (k0 + 32 < Kd)
          __builtin_prefetch((const void*)(A + sr * (long long)Kd + k0 + 32), 0, 0);
      }
      lA[row * 4 + kc] = va;
    }

    if (w == 0) __builtin_amdgcn_s_wait_tensorcnt(0);  // B tile landed in LDS
    __syncthreads();

    // ---- fragments per ISA bf16 layouts (A 16x32, B 32x16)
    FragAB fa[2], fb[4];
#pragma unroll
    for (int mt = 0; mt < 2; ++mt) {
      const int r = wm * 32 + mt * 16 + lm;
      fa[mt].u[0] = lA[r * 4 + lh];          // K = lh*8 .. +7
      fa[mt].u[1] = lA[r * 4 + 2 + lh];      // K = 16 + lh*8 .. +7
    }
#pragma unroll
    for (int nt = 0; nt < 4; ++nt) {
      const int r = wn * 64 + nt * 16 + lm;
      fb[nt].u[0] = lB[r * 4 + lh];          // K = lh*8 .. +7
      fb[nt].u[1] = lB[r * 4 + 2 + lh];      // K = 16 + lh*8 .. +7
    }
#pragma unroll
    for (int mt = 0; mt < 2; ++mt)
#pragma unroll
      for (int nt = 0; nt < 4; ++nt)
        acc[mt][nt] = __builtin_amdgcn_wmma_f32_16x16x32_bf16(
            false, fa[mt].v, false, fb[nt].v, (short)0, acc[mt][nt], false, false);
    __syncthreads();
  }

  // ---- epilogue: C/D f32 layout (lane n = lm, row m = v + lh*8)
#pragma unroll
  for (int mt = 0; mt < 2; ++mt) {
#pragma unroll
    for (int nt = 0; nt < 4; ++nt) {
#pragma unroll
      for (int v = 0; v < 8; ++v) {
        const long long R = mBase + wm * 32 + mt * 16 + lh * 8 + v;
        const long long C = nBase + wn * 64 + nt * 16 + lm;
        const long long idx = R * (long long)N + C;
        float a = acc[mt][nt][v];
        float val = a + (bias ? bias[C] : 0.0f);
        switch (mode) {
          case MODE_F32:     outF[idx] = val; break;
          case MODE_ACCUM:   outF[idx] += val; break;
          case MODE_ACCUM_LAST: {
            float t2 = outF[idx] + val;
            outF[idx] = t2;
            outB[idx] = f2bf(t2);
          } break;
          case MODE_BF16:    outB[idx] = f2bf(val); break;
          case MODE_BOTH:    outF[idx] = val; outB[idx] = f2bf(val); break;
          case MODE_SIGMOID: outF[idx] = 1.0f / (1.0f + expf(-val)); break;
          case MODE_GELUMUL: outB[idx] = f2bf(gelu_exact(val) * val); break;
          case MODE_RES:     outF[idx] = val + res[idx]; break;
        }
      }
    }
  }
}

// ---------------------------------------------------------------------------
// RG-LRU sequential scan. r/i already sigmoided by the gate GEMMs.
// One thread per (b, d): 4096 threads, coalesced over d.
// ---------------------------------------------------------------------------
__global__ __launch_bounds__(256) void k_scan(const float* __restrict__ r,
                                              const float* __restrict__ i,
                                              const float* __restrict__ xt,
                                              const float* __restrict__ lam,
                                              float* __restrict__ h) {
  const int idx = blockIdx.x * blockDim.x + threadIdx.x;
  if (idx >= NBATCH * DMODEL) return;
  const int b = idx >> 9, d = idx & (DMODEL - 1);
  const float coef = -8.0f * log1pf(expf(lam[d]));    // -8 * softplus(lam)
  float hv = 0.0f;
  const long long base = (long long)b * SEQ * DMODEL + d;
  for (int s = 0; s < SEQ; ++s) {
    const long long o = base + (long long)s * DMODEL;
    const float la = coef * r[o];
    const float av = expf(la);
    const float bb = sqrtf(fmaxf(0.0f, 1.0f - expf(2.0f * la))) * (i[o] * xt[o]);
    hv = av * hv + bb;
    h[o] = hv;
  }
}

// x1 = h * gelu(l2) + skip
__global__ __launch_bounds__(256) void k_merge(const float* __restrict__ h,
                                               const float* __restrict__ l2,
                                               const float* __restrict__ skip,
                                               float* __restrict__ x1) {
  const long long idx = (long long)blockIdx.x * blockDim.x + threadIdx.x;
  if (idx < (long long)NTOK * DMODEL)
    x1[idx] = h[idx] * gelu_exact(l2[idx]) + skip[idx];
}

// ---------------------------------------------------------------------------
extern "C" void kernel_launch(void* const* d_in, const int* in_sizes, int n_in,
                              void* d_out, int out_size, void* d_ws, size_t ws_size,
                              hipStream_t stream) {
  (void)in_sizes; (void)n_in; (void)ws_size; (void)out_size;
  const float* x    = (const float*)d_in[0];
  const float* g    = (const float*)d_in[1];
  const float* W1   = (const float*)d_in[2];
  const float* b1   = (const float*)d_in[3];
  const float* W2   = (const float*)d_in[4];
  const float* b2   = (const float*)d_in[5];
  const float* dw_w = (const float*)d_in[6];
  const float* dw_b = (const float*)d_in[7];
  const float* pw_w = (const float*)d_in[8];
  const float* pw_b = (const float*)d_in[9];
  const float* Wi   = (const float*)d_in[10];
  const float* bi   = (const float*)d_in[11];
  const float* Wr   = (const float*)d_in[12];
  const float* br   = (const float*)d_in[13];
  const float* lam  = (const float*)d_in[14];
  const float* Wm1  = (const float*)d_in[15];
  const float* bm1  = (const float*)d_in[16];
  const float* Wm2  = (const float*)d_in[17];
  const float* bm2  = (const float*)d_in[18];
  float* out = (float*)d_out;

  // workspace carve-out (256B aligned)
  char* w = (char*)d_ws;
  size_t off = 0;
  auto alloc = [&](size_t bytes) -> char* {
    char* p = w + off;
    off = (off + bytes + 255) & ~(size_t)255;
    return p;
  };
  typedef unsigned short u16;
  const size_t DD = (size_t)DMODEL * DMODEL;
  u16* W1T  = (u16*)alloc(DD * 2);
  u16* W2T  = (u16*)alloc(DD * 2);
  u16* WiT  = (u16*)alloc(DD * 2);
  u16* WrT  = (u16*)alloc(DD * 2);
  u16* pwT  = (u16*)alloc(DD * 2);
  u16* Wm1T = (u16*)alloc((size_t)DMID * DMODEL * 2);
  u16* Wm2T = (u16*)alloc((size_t)DMODEL * DMID * 2);
  u16* dwT  = (u16*)alloc((size_t)KCONV * DD * 2);
  u16* xnb   = (u16*)alloc((size_t)NTOK * DMODEL * 2);
  u16* l1b   = (u16*)alloc((size_t)NTOK * DMODEL * 2);
  float* l2f = (float*)alloc((size_t)NTOK * DMODEL * 4);
  float* c1f = (float*)alloc((size_t)NTOK * DMODEL * 4);
  u16* c1b   = (u16*)alloc((size_t)NTOK * DMODEL * 2);
  float* c2f = (float*)alloc((size_t)NTOK * DMODEL * 4);
  u16* c2b   = (u16*)alloc((size_t)NTOK * DMODEL * 2);
  float* rf  = (float*)alloc((size_t)NTOK * DMODEL * 4);
  float* iff = (float*)alloc((size_t)NTOK * DMODEL * 4);
  float* hf  = (float*)alloc((size_t)NTOK * DMODEL * 4);
  float* x1f = (float*)alloc((size_t)NTOK * DMODEL * 4);
  u16* xn2b  = (u16*)alloc((size_t)NTOK * DMODEL * 2);
  u16* hmidb = (u16*)alloc((size_t)NTOK * DMID * 2);

  const dim3 blk(256);
  auto cvt = [&](const float* s, u16* dst, int N, int K, long long sN, long long sK,
                 long long o) {
    long long total = (long long)N * K;
    int nb = (int)((total + 255) / 256);
    k_cvt_t<<<nb, blk, 0, stream>>>(s, dst, N, K, sN, sK, o);
  };
  // weights -> bf16 [N][K] (B^T) layout
  cvt(W1,  W1T,  DMODEL, DMODEL, 1, DMODEL, 0);
  cvt(W2,  W2T,  DMODEL, DMODEL, 1, DMODEL, 0);
  cvt(Wi,  WiT,  DMODEL, DMODEL, 1, DMODEL, 0);
  cvt(Wr,  WrT,  DMODEL, DMODEL, 1, DMODEL, 0);
  cvt(pw_w, pwT, DMODEL, DMODEL, DMODEL, 1, 0);               // w[o,i,0]
  cvt(Wm1, Wm1T, DMID,   DMODEL, 1, DMID, 0);
  cvt(Wm2, Wm2T, DMODEL, DMID,   1, DMODEL, 0);
  for (int kk = 0; kk < KCONV; ++kk)                          // w[o,i,kk]
    cvt(dw_w, dwT + (size_t)kk * DD, DMODEL, DMODEL,
        (long long)DMODEL * KCONV, KCONV, kk);

  auto gemm = [&](const u16* A, const u16* BT, const float* bias, float* oF,
                  u16* oB, const float* res, int M, int N, int K, int mode,
                  int shift, int applyShift) {
    dim3 grid(N / 128, M / 128);
    k_gemm<<<grid, blk, 0, stream>>>(A, BT, bias, oF, oB, res, M, N, K, mode,
                                     shift, applyShift);
  };

  // --- branch 1 ---
  k_rms<<<NTOK, blk, 0, stream>>>(x, g, xnb);
  gemm(xnb, W1T, b1, nullptr, l1b, nullptr, NTOK, DMODEL, DMODEL, MODE_BF16, 0, 0);
  gemm(xnb, W2T, b2, l2f, nullptr, nullptr, NTOK, DMODEL, DMODEL, MODE_F32, 0, 0);
  // dense K=9 'same' conv over channels as 9 shifted accumulated GEMMs
  for (int kk = 0; kk < KCONV; ++kk) {
    int mode = (kk == 0) ? MODE_F32 : (kk == KCONV - 1 ? MODE_ACCUM_LAST : MODE_ACCUM);
    gemm(l1b, dwT + (size_t)kk * DD, (kk == 0) ? dw_b : nullptr, c1f,
         (kk == KCONV - 1) ? c1b : nullptr, nullptr, NTOK, DMODEL, DMODEL,
         mode, kk - KCONV / 2, 1);
  }
  gemm(c1b, pwT, pw_b, c2f, c2b, nullptr, NTOK, DMODEL, DMODEL, MODE_BOTH, 0, 0);
  // gates (sigmoid applied in epilogue)
  gemm(c2b, WrT, br, rf,  nullptr, nullptr, NTOK, DMODEL, DMODEL, MODE_SIGMOID, 0, 0);
  gemm(c2b, WiT, bi, iff, nullptr, nullptr, NTOK, DMODEL, DMODEL, MODE_SIGMOID, 0, 0);
  k_scan<<<(NBATCH * DMODEL + 255) / 256, blk, 0, stream>>>(rf, iff, c2f, lam, hf);
  k_merge<<<(int)(((long long)NTOK * DMODEL + 255) / 256), blk, 0, stream>>>(hf, l2f, x, x1f);

  // --- branch 2 (MLP) ---
  k_rms<<<NTOK, blk, 0, stream>>>(x1f, g, xn2b);
  gemm(xn2b, Wm1T, bm1, nullptr, hmidb, nullptr, NTOK, DMID, DMODEL, MODE_GELUMUL, 0, 0);
  gemm(hmidb, Wm2T, bm2, out, nullptr, x1f, NTOK, DMODEL, DMID, MODE_RES, 0, 0);
}
